// ImprovedGNNModel_21328807592521
// MI455X (gfx1250) — matrix operand order repaired and checked by reading, not compiled, not run on hardware
//
#include <hip/hip_runtime.h>
#include <hip/hip_bf16.h>

#define F_IN   128
#define H_CH   64
#define HEADS  8
#define C_OUT  512   // HEADS * H_CH

typedef __bf16 bf16_t;
typedef bf16_t v16bf __attribute__((ext_vector_type(16)));
typedef bf16_t v8bf  __attribute__((ext_vector_type(8)));
typedef float  v8f   __attribute__((ext_vector_type(8)));

// ---------------------------------------------------------------- small utils
__device__ inline void atomicMaxFloat(float* addr, float val) {
    if (val >= 0.0f) atomicMax((int*)addr, __float_as_int(val));
    else             atomicMin((unsigned int*)addr, __float_as_uint(val));
}

// ---------------------------------------------------------------- param MLP
// p = relu(uw @ Wp1 + bp1) @ Wp2 + bp2   -> [64]
__global__ void param_mlp_kernel(const float* __restrict__ uw,
                                 const float* __restrict__ Wp1,
                                 const float* __restrict__ bp1,
                                 const float* __restrict__ Wp2,
                                 const float* __restrict__ bp2,
                                 float* __restrict__ p) {
    __shared__ float hid[H_CH];
    int t = threadIdx.x;                       // 64 threads
    float h = uw[0] * Wp1[t] + uw[1] * Wp1[H_CH + t] + bp1[t];
    hid[t] = fmaxf(h, 0.0f);
    __syncthreads();
    float s = bp2[t];
    #pragma unroll 8
    for (int k = 0; k < H_CH; ++k) s += hid[k] * Wp2[k * H_CH + t];
    p[t] = s;
}

// c1[j] = sum_k p[k] * W1[(128+k)*512 + j]          (pe contribution, layer 1)
__global__ void const_row1_kernel(const float* __restrict__ W1,
                                  const float* __restrict__ p,
                                  float* __restrict__ c1) {
    int j = blockIdx.x * blockDim.x + threadIdx.x;
    if (j >= C_OUT) return;
    float s = 0.0f;
    for (int k = 0; k < H_CH; ++k) s += p[k] * W1[(size_t)(F_IN + k) * C_OUT + j];
    c1[j] = s;
}

// c2[j] = sum_k p[k&63] * W2[(512+k)*512 + j]       (tile(pe,8) contribution, layer 2)
__global__ void const_row2_kernel(const float* __restrict__ W2,
                                  const float* __restrict__ p,
                                  float* __restrict__ c2) {
    int j = blockIdx.x * blockDim.x + threadIdx.x;
    if (j >= C_OUT) return;
    float s = 0.0f;
    for (int k = 0; k < C_OUT; ++k) s += p[k & 63] * W2[(size_t)(C_OUT + k) * C_OUT + j];
    c2[j] = s;
}

// ---------------------------------------------------------------- weight swizzle
// Pack W[0:K, 0:512] (f32 row-major) into bf16 WMMA B-fragment order:
//   out[(((nt*(K/32)+kt)*32 + lane)*16 + e)] = W[kt*32 + (lane>>4)*8 + (e&7) + (e>>3)*16][nt*16 + (lane&15)]
__global__ void swizzle_b_kernel(const float* __restrict__ W, bf16_t* __restrict__ out,
                                 int K) {
    size_t idx = (size_t)blockIdx.x * blockDim.x + threadIdx.x;
    size_t total = (size_t)K * C_OUT;
    if (idx >= total) return;
    int e    = (int)(idx & 15);
    int lane = (int)((idx >> 4) & 31);
    size_t rest = idx >> 9;
    int kt = (int)(rest % (K / 32));
    int nt = (int)(rest / (K / 32));
    int k   = kt * 32 + (lane >> 4) * 8 + (e & 7) + (e >> 3) * 16;
    int col = nt * 16 + (lane & 15);
    out[idx] = (bf16_t)W[(size_t)k * C_OUT + col];
}

// ---------------------------------------------------------------- WMMA GEMM
// H[n, 0:512] = X[n, 0:K] @ Wsw  + cconst[col]
// Block: 256 threads (8 waves). Block computes 16 rows x 512 cols; wave w owns cols [w*64, w*64+64).
// A staged through LDS as bf16 (shared by all 8 waves), B read pre-swizzled from global.
template<int K>
__global__ __launch_bounds__(256)
void gemm_wmma_kernel(const float* __restrict__ X,
                      const bf16_t* __restrict__ Bsw,
                      const float* __restrict__ cconst,
                      float* __restrict__ Hout,
                      int nrows) {
    constexpr int LDA = K + 8;                 // bf16 elems; +16B pad vs bank conflicts
    __shared__ bf16_t Alds[16 * LDA];

    const int tid  = threadIdx.x;
    const int rowbase = blockIdx.x * 16;

    // cooperative load of the 16xK f32 A tile -> bf16 LDS
    for (int idx4 = tid; idx4 < 16 * (K / 4); idx4 += 256) {
        int r  = idx4 / (K / 4);
        int kq = idx4 % (K / 4);
        float4 v = make_float4(0.f, 0.f, 0.f, 0.f);
        int row = rowbase + r;
        if (row < nrows)
            v = *(const float4*)(X + (size_t)row * K + kq * 4);
        bf16_t* dst = &Alds[r * LDA + kq * 4];
        dst[0] = (bf16_t)v.x; dst[1] = (bf16_t)v.y;
        dst[2] = (bf16_t)v.z; dst[3] = (bf16_t)v.w;
    }
    __syncthreads();

    const int wave = tid >> 5;
    const int lane = tid & 31;
    const int m    = lane & 15;
    const int hi   = lane >> 4;
    const int colbase = wave * 64;

    v8f acc[4];
    #pragma unroll
    for (int t = 0; t < 4; ++t) acc[t] = (v8f){0.f,0.f,0.f,0.f,0.f,0.f,0.f,0.f};

    const bf16_t* arow = &Alds[m * LDA];

    #pragma unroll
    for (int kt = 0; kt < K / 32; ++kt) {
        const int k0 = kt * 32 + hi * 8;
        v8bf alo = *(const v8bf*)(arow + k0);
        v8bf ahi = *(const v8bf*)(arow + k0 + 16);
        v16bf afrag = __builtin_shufflevector(alo, ahi,
            0,1,2,3,4,5,6,7,8,9,10,11,12,13,14,15);
        #pragma unroll
        for (int t = 0; t < 4; ++t) {
            const int nt = (colbase >> 4) + t;
            const bf16_t* bp = Bsw + ((size_t)((nt * (K / 32) + kt) * 32 + lane) << 4);
            v8bf blo = *(const v8bf*)bp;
            v8bf bhi = *(const v8bf*)(bp + 8);
            v16bf bfrag = __builtin_shufflevector(blo, bhi,
                0,1,2,3,4,5,6,7,8,9,10,11,12,13,14,15);
            acc[t] = __builtin_amdgcn_wmma_f32_16x16x32_bf16(
                false, afrag, false, bfrag, (short)0, acc[t], false, false);
        }
    }

    // D layout: col n = lane&15 ; VGPR v -> row (lane>>4)*8 + v
    const int n  = lane & 15;
    const int mo = (lane >> 4) * 8;
    #pragma unroll
    for (int t = 0; t < 4; ++t) {
        int col = colbase + t * 16 + n;
        float cc = cconst[col];
        #pragma unroll
        for (int v = 0; v < 8; ++v) {
            int row = rowbase + mo + v;
            if (row < nrows)
                Hout[(size_t)row * C_OUT + col] = acc[t][v] + cc;
        }
    }
}

// ---------------------------------------------------------------- attention coeffs
// a_src[n,h] = <h[n,h,:], as[h,:]> ; a_dst likewise
__global__ void attn_coeff_kernel(const float* __restrict__ H,
                                  const float* __restrict__ as_,
                                  const float* __restrict__ ad_,
                                  float* __restrict__ a_src,
                                  float* __restrict__ a_dst, int n) {
    size_t id = (size_t)blockIdx.x * blockDim.x + threadIdx.x;
    if (id >= (size_t)n * HEADS) return;
    int node = (int)(id >> 3), hd = (int)(id & 7);
    const float4* hv  = (const float4*)(H + (size_t)node * C_OUT + hd * H_CH);
    const float4* av  = (const float4*)(as_ + hd * H_CH);
    const float4* bv  = (const float4*)(ad_ + hd * H_CH);
    float ss = 0.f, sd = 0.f;
    #pragma unroll
    for (int q = 0; q < H_CH / 4; ++q) {
        float4 h4 = hv[q], a4 = av[q], b4 = bv[q];
        ss += h4.x*a4.x + h4.y*a4.y + h4.z*a4.z + h4.w*a4.w;
        sd += h4.x*b4.x + h4.y*b4.y + h4.z*b4.z + h4.w*b4.w;
    }
    a_src[id] = ss; a_dst[id] = sd;
}

// ---------------------------------------------------------------- per-layer init
__global__ void init_layer_kernel(float* __restrict__ outacc, float* __restrict__ m,
                                  float* __restrict__ den, float* __restrict__ pooled,
                                  int zero_pooled, size_t n512, size_t n8) {
    size_t i = (size_t)blockIdx.x * blockDim.x + threadIdx.x;
    size_t stride = (size_t)gridDim.x * blockDim.x;
    for (; i < n512; i += stride) {
        outacc[i] = 0.0f;
        if (i < n8) { m[i] = -__builtin_inff(); den[i] = 0.0f; }
        if (zero_pooled && i < C_OUT) pooled[i] = 0.0f;
    }
}

// ---------------------------------------------------------------- edge pass A: segment max
__global__ void edge_max_kernel(const int* __restrict__ ei, int E, int n,
                                const float* __restrict__ a_src,
                                const float* __restrict__ a_dst,
                                float* __restrict__ m) {
    size_t e = (size_t)blockIdx.x * blockDim.x + threadIdx.x;
    size_t Etot = (size_t)E + n;
    if (e >= Etot) return;
    int s, d;
    if (e < (size_t)E) { s = ei[e]; d = ei[(size_t)E + e]; }
    else               { s = d = (int)(e - E); }
    const float* As = a_src + (size_t)s * HEADS;
    const float* Ad = a_dst + (size_t)d * HEADS;
    #pragma unroll
    for (int h = 0; h < HEADS; ++h) {
        float v = As[h] + Ad[h];
        v = (v > 0.f) ? v : 0.2f * v;          // leaky_relu(0.2)
        atomicMaxFloat(&m[(size_t)d * HEADS + h], v);
    }
}

// ---------------------------------------------------------------- edge pass B: fused softmax-weighted scatter
// wave per edge: lane owns 16 contiguous channels (one head per lane group of 4).
// Accumulates denom[dst,h] += e^x and out[dst,:] += e^x * h[src,:]; alpha normalized later.
__global__ __launch_bounds__(256)
void edge_accum_kernel(const int* __restrict__ ei, int E, int n,
                       const float* __restrict__ a_src,
                       const float* __restrict__ a_dst,
                       const float* __restrict__ m,
                       const float* __restrict__ H,
                       float* __restrict__ den,
                       float* __restrict__ outacc) {
    const int lane = threadIdx.x & 31;
    const int head = lane >> 2;
    const int c0   = lane * 16;
    size_t wid = (size_t)blockIdx.x * (blockDim.x >> 5) + (threadIdx.x >> 5);
    size_t nw  = (size_t)gridDim.x * (blockDim.x >> 5);
    size_t Etot = (size_t)E + n;
    for (size_t e = wid; e < Etot; e += nw) {
        int s, d;
        if (e < (size_t)E) { s = ei[e]; d = ei[(size_t)E + e]; }
        else               { s = d = (int)(e - E); }
        float av = a_src[(size_t)s * HEADS + head] + a_dst[(size_t)d * HEADS + head];
        av = (av > 0.f) ? av : 0.2f * av;
        float ex = __expf(av - m[(size_t)d * HEADS + head]);
        if ((lane & 3) == 0) atomicAdd(&den[(size_t)d * HEADS + head], ex);
        const float4* hp = (const float4*)(H + (size_t)s * C_OUT + c0);
        float* op = outacc + (size_t)d * C_OUT + c0;
        #pragma unroll
        for (int j = 0; j < 4; ++j) {
            float4 hv = hp[j];
            atomicAdd(op + j * 4 + 0, ex * hv.x);
            atomicAdd(op + j * 4 + 1, ex * hv.y);
            atomicAdd(op + j * 4 + 2, ex * hv.z);
            atomicAdd(op + j * 4 + 3, ex * hv.w);
        }
    }
}

// ---------------------------------------------------------------- node epilogue
// out = ELU(outacc/denom + bias); optional store + optional mean-pool accumulate
__global__ __launch_bounds__(512)
void node_epilogue_kernel(const float* __restrict__ outacc,
                          const float* __restrict__ den,
                          const float* __restrict__ bias,
                          float* __restrict__ out,     // may be nullptr
                          float* __restrict__ pooled,
                          int n, int do_pool) {
    const int c = threadIdx.x;                 // 512 channels
    const int hd = c >> 6;
    float local = 0.0f;
    for (int node = blockIdx.x; node < n; node += gridDim.x) {
        float v = outacc[(size_t)node * C_OUT + c] / den[(size_t)node * HEADS + hd] + bias[c];
        v = (v > 0.f) ? v : (__expf(v) - 1.0f); // ELU
        if (out) out[(size_t)node * C_OUT + c] = v;
        local += v;
    }
    if (do_pool) atomicAdd(&pooled[c], local);
}

// ---------------------------------------------------------------- final head
__global__ void final_kernel(const float* __restrict__ pooled,
                             const float* __restrict__ p,
                             const float* __restrict__ Wl,
                             const float* __restrict__ bl,
                             float* __restrict__ out, int n) {
    __shared__ float red[256];
    int t = threadIdx.x;
    float s = 0.0f;
    float invn = 1.0f / (float)n;
    for (int i = t; i < C_OUT + H_CH; i += 256) {
        float f = (i < C_OUT) ? pooled[i] * invn : p[i - C_OUT];
        s += f * Wl[i];
    }
    red[t] = s; __syncthreads();
    for (int off = 128; off > 0; off >>= 1) {
        if (t < off) red[t] += red[t + off];
        __syncthreads();
    }
    if (t == 0) out[0] = red[0] + bl[0];
}

// ---------------------------------------------------------------- launch
extern "C" void kernel_launch(void* const* d_in, const int* in_sizes, int n_in,
                              void* d_out, int out_size, void* d_ws, size_t ws_size,
                              hipStream_t stream) {
    const float* x   = (const float*)d_in[0];
    const int*   ei  = (const int*)  d_in[1];
    const float* uw  = (const float*)d_in[2];
    const float* Wp1 = (const float*)d_in[3];
    const float* bp1 = (const float*)d_in[4];
    const float* Wp2 = (const float*)d_in[5];
    const float* bp2 = (const float*)d_in[6];
    const float* W1  = (const float*)d_in[7];
    const float* as1 = (const float*)d_in[8];
    const float* ad1 = (const float*)d_in[9];
    const float* b1  = (const float*)d_in[10];
    const float* W2  = (const float*)d_in[11];
    const float* as2 = (const float*)d_in[12];
    const float* ad2 = (const float*)d_in[13];
    const float* b2  = (const float*)d_in[14];
    const float* Wl  = (const float*)d_in[15];
    const float* bl  = (const float*)d_in[16];
    float* out = (float*)d_out;

    const int N = in_sizes[0] / F_IN;
    const int E = in_sizes[1] / 2;
    const size_t N8 = (size_t)N * HEADS;
    const size_t NC = (size_t)N * C_OUT;
    const size_t Etot = (size_t)E + N;

    // workspace carve-up (floats)
    float* ws = (float*)d_ws;
    size_t off = 0;
    float* P    = ws + off; off += 64;
    float* C1   = ws + off; off += C_OUT;
    float* C2   = ws + off; off += C_OUT;
    float* POOL = ws + off; off += C_OUT;
    off = (off + 63) & ~(size_t)63;
    float* ASRC = ws + off; off += N8;
    float* ADST = ws + off; off += N8;
    float* MBUF = ws + off; off += N8;
    float* DEN  = ws + off; off += N8;
    off = (off + 63) & ~(size_t)63;
    float* BUFA = ws + off; off += NC;   // h (GEMM out / gather source)
    float* BUFB = ws + off; off += NC;   // ELU(layer1) = layer2 GEMM input
    float* BUFC = ws + off; off += NC;   // edge accumulator
    bf16_t* W1SW = (bf16_t*)(ws + off); off += (size_t)F_IN  * C_OUT / 2;
    bf16_t* W2SW = (bf16_t*)(ws + off); off += (size_t)C_OUT * C_OUT / 2;
    (void)ws_size; (void)n_in; (void)out_size;

    const int gridM = (N + 15) / 16;
    const int initBlocks = 8192;
    const int edgeBlocksA = (int)((Etot + 255) / 256);

    // prep: tiny MLP, folded constant rows, weight swizzles
    param_mlp_kernel<<<1, 64, 0, stream>>>(uw, Wp1, bp1, Wp2, bp2, P);
    const_row1_kernel<<<2, 256, 0, stream>>>(W1, P, C1);
    const_row2_kernel<<<2, 256, 0, stream>>>(W2, P, C2);
    swizzle_b_kernel<<<(F_IN * C_OUT + 255) / 256, 256, 0, stream>>>(W1, W1SW, F_IN);
    swizzle_b_kernel<<<(C_OUT * C_OUT + 255) / 256, 256, 0, stream>>>(W2, W2SW, C_OUT);

    // ---- layer 1
    gemm_wmma_kernel<F_IN><<<gridM, 256, 0, stream>>>(x, W1SW, C1, BUFA, N);
    attn_coeff_kernel<<<(int)((N8 + 255) / 256), 256, 0, stream>>>(BUFA, as1, ad1, ASRC, ADST, N);
    init_layer_kernel<<<initBlocks, 256, 0, stream>>>(BUFC, MBUF, DEN, POOL, 0, NC, N8);
    edge_max_kernel<<<edgeBlocksA, 256, 0, stream>>>(ei, E, N, ASRC, ADST, MBUF);
    edge_accum_kernel<<<4096, 256, 0, stream>>>(ei, E, N, ASRC, ADST, MBUF, BUFA, DEN, BUFC);
    node_epilogue_kernel<<<512, 512, 0, stream>>>(BUFC, DEN, b1, BUFB, POOL, N, 0);

    // ---- layer 2
    gemm_wmma_kernel<C_OUT><<<gridM, 256, 0, stream>>>(BUFB, W2SW, C2, BUFA, N);
    attn_coeff_kernel<<<(int)((N8 + 255) / 256), 256, 0, stream>>>(BUFA, as2, ad2, ASRC, ADST, N);
    init_layer_kernel<<<initBlocks, 256, 0, stream>>>(BUFC, MBUF, DEN, POOL, 1, NC, N8);
    edge_max_kernel<<<edgeBlocksA, 256, 0, stream>>>(ei, E, N, ASRC, ADST, MBUF);
    edge_accum_kernel<<<4096, 256, 0, stream>>>(ei, E, N, ASRC, ADST, MBUF, BUFA, DEN, BUFC);
    node_epilogue_kernel<<<512, 512, 0, stream>>>(BUFC, DEN, b2, nullptr, POOL, N, 1);

    // ---- head
    final_kernel<<<1, 256, 0, stream>>>(POOL, P, Wl, bl, out, N);
}
